// Model_71442486001934
// MI455X (gfx1250) — matrix operand order repaired and checked by reading, not compiled, not run on hardware
//
#include <hip/hip_runtime.h>
#include <hip/hip_bf16.h>

typedef __attribute__((ext_vector_type(16))) __bf16 v16bf;
typedef __attribute__((ext_vector_type(8)))  float  v8f;

#define KBN 0.9999950000374997f   // 1/sqrt(1+1e-5), eval-mode BN
#define WFULL 251
#define LDSK 40                   // LDS row stride (elems): 80B, keeps b128 alignment

__device__ __forceinline__ float elu1(float x) { return x > 0.f ? x : (expf(x) - 1.f); }
__device__ __forceinline__ int imin(int a, int b) { return a < b ? a : b; }
__device__ __forceinline__ int imax(int a, int b) { return a > b ? a : b; }

// ---------------------------------------------------------------------------
// Generic bf16 WMMA GEMM: C[M,N] = op(A)[M,K] * B[K,N], f32 accumulate.
// Block tile 64(M) x 128(N), 256 threads = 8 waves (4 in M x 2 in N),
// each wave computes 16x64 via 4 WMMA accumulators per 32-deep K step.
// BMODE 0: B dense row-major (ldb); A loaded branchless (clamped + select).
// BMODE 1: B = implicit im2col gather from NCHW bf16 activation; A (weights,
//          zero-padded buffer) streamed with one global_load_async_to_lds_b128
//          per thread per K-step (ASYNCcnt-tracked global->LDS DMA).
// ATRANS: A[m,k] = A[k*lda + m] (channel-attention matrix).
// Epilogue: +bias -> BN -> ELU -> +residual, optional bf16 mirror.
// ---------------------------------------------------------------------------
template<int BMODE, bool ATRANS>
__global__ __launch_bounds__(256) void gemm_bf16(
    const __bf16* __restrict__ A, const __bf16* __restrict__ Bm,
    float* __restrict__ C, __bf16* __restrict__ Cbf,
    int M, int N, int K, int lda, int ldb,
    long sAb, long sBb, long sCb, int npi,
    const float* __restrict__ bias, const float* __restrict__ bnG,
    const float* __restrict__ bnB, int doElu, const float* __restrict__ res,
    int Cin, int H, int W, int Ho, int Wo, int KH, int KW, int SH, int PH, int PW)
{
  __shared__ __bf16 sA[64][LDSK];
  __shared__ __bf16 sB[128][LDSK];
  const int t  = threadIdx.x;
  const int bz = blockIdx.z;
  const int n0 = blockIdx.x * 128;
  const int m0 = blockIdx.y * 64;
  const __bf16* Ab = A + (long)bz * sAb;
  const __bf16* Bb = Bm + (long)bz * sBb;
  const int wv = t >> 5, ln = t & 31;
  const int wm = wv & 3, wn = wv >> 2;      // 4 waves in M, 2 in N
  const int fm = ln & 15, hi = ln >> 4;     // lane -> fragment row / half

  // --- loop-invariant per-thread loader coordinates: row = t&63 (lanes span
  // consecutive rows -> coalesced along N), 8 consecutive k per thread.
  const int lr  = t & 63;
  const int lkb = (t >> 6) << 3;            // 0,8,16,24

  // B-side invariants (two row-groups: lr and lr+64)
  int  gncv[2]; bool nval[2];
  int  bb[2], oh[2], ow[2];
#pragma unroll
  for (int g = 0; g < 2; ++g) {
    int gn = n0 + lr + 64 * g;
    nval[g] = gn < N;
    int gnc = imin(gn, N - 1);
    gncv[g] = gnc;
    if (BMODE == 1) {
      const int hw = Ho * Wo;
      bb[g] = gnc / hw; int pos = gnc - bb[g] * hw;
      oh[g] = pos / Wo; ow[g] = pos - oh[g] * Wo;
    } else {
      bb[g] = oh[g] = ow[g] = 0;
    }
  }
  // A-side invariants (dense path)
  const int gm_l = m0 + lr;
  const int gmc  = imin(gm_l, M - 1);
  const bool mvalid = gm_l < M;

  v8f acc[4];
#pragma unroll
  for (int s = 0; s < 4; ++s) acc[s] = (v8f){0.f,0.f,0.f,0.f,0.f,0.f,0.f,0.f};

  for (int k0 = 0; k0 < K; k0 += 32) {
    // ---- A tile (64x32) ----
    if (BMODE == 1) {
      // one 16B async DMA per thread: weights buffer is zero-padded so the
      // unguarded reads stay in-allocation; B's k-tail zeros kill any overrun.
      unsigned loff = (unsigned)(unsigned long long)&sA[lr][lkb];
      unsigned long long ga =
          (unsigned long long)(Ab + (long)(m0 + lr) * lda + (k0 + lkb));
      asm volatile("global_load_async_to_lds_b128 %0, %1, off"
                   :: "v"(loff), "v"(ga) : "memory");
    } else {
      union { __bf16 h[8]; uint4 q; } ua;
#pragma unroll
      for (int j = 0; j < 8; ++j) {
        int gk  = k0 + lkb + j;
        int gkc = imin(gk, K - 1);
        __bf16 bv = ATRANS ? Ab[(long)gkc * lda + gmc]
                           : Ab[(long)gmc * lda + gkc];
        ua.h[j] = (mvalid && gk < K) ? bv : (__bf16)0.f;
      }
      *(uint4*)&sA[lr][lkb] = ua.q;
    }

    // ---- B tile (32x128, stored transposed [n][k]) ----
#pragma unroll
    for (int g = 0; g < 2; ++g) {
      union { __bf16 h[8]; uint4 q; } ub;
#pragma unroll
      for (int j = 0; j < 8; ++j) {
        int gk  = k0 + lkb + j;
        int gkc = imin(gk, K - 1);
        bool v  = nval[g] && (gk < K);
        __bf16 bv;
        if (BMODE == 0) {
          bv = Bb[(long)gkc * ldb + gncv[g]];
        } else {
          const int khw = KH * KW;
          int ci = gkc / khw; int rem = gkc - ci * khw;
          int kh = rem / KW,  kw = rem - kh * KW;
          int ih = oh[g] * SH + kh - PH;
          int iw = ow[g] + kw - PW;
          v = v && ((unsigned)ih < (unsigned)H) && ((unsigned)iw < (unsigned)W);
          int ihc = imin(imax(ih, 0), H - 1);
          int iwc = imin(imax(iw, 0), W - 1);
          bv = Bb[(((long)(bb[g] * Cin + ci)) * H + ihc) * W + iwc];
        }
        ub.h[j] = v ? bv : (__bf16)0.f;
      }
      *(uint4*)&sB[lr + 64 * g][lkb] = ub.q;
    }
    if (BMODE == 0 && k0 + 32 < K)
      __builtin_prefetch(Bb + (long)imin(k0 + 32 + lkb, K - 1) * ldb + gncv[0], 0, 0);

    if (BMODE == 1)
      asm volatile("s_wait_asynccnt 0x0" ::: "memory");
    __syncthreads();

    // ---- fragments per ISA 16-bit A 16x32 / B 32x16 layouts ----
    union { unsigned u[8]; v16bf v; } fa;
    const __bf16* arow = &sA[wm * 16 + fm][0];
#pragma unroll
    for (int r = 0; r < 8; ++r) {
      int ka = ((r >> 2) << 4) + (hi << 3) + ((r & 3) << 1);  // A: K pair base
      fa.u[r] = *(const unsigned*)(arow + ka);
    }
#pragma unroll
    for (int s = 0; s < 4; ++s) {
      union { unsigned u[8]; v16bf v; } fb;
      const __bf16* brow = &sB[wn * 64 + s * 16 + fm][0];
#pragma unroll
      for (int r = 0; r < 8; ++r) {
        int kb = ((r >> 2) << 4) + (hi << 3) + ((r & 3) << 1);  // B: K pair base
        fb.u[r] = *(const unsigned*)(brow + kb);
      }
      acc[s] = __builtin_amdgcn_wmma_f32_16x16x32_bf16(false, fa.v, false, fb.v,
                                                       (short)0, acc[s], false, false);
    }
    __syncthreads();
  }

  // --- epilogue (C/D layout: VGPR r, lane -> M = r + 8*hi, N = lane&15)
#pragma unroll
  for (int sub = 0; sub < 4; ++sub) {
    int gn = n0 + wn * 64 + sub * 16 + fm;
    if (gn >= N) continue;
    int img = gn / npi; int pos = gn - img * npi;
#pragma unroll
    for (int r = 0; r < 8; ++r) {
      int gm = m0 + wm * 16 + hi * 8 + r;
      if (gm >= M) continue;
      float y = acc[sub][r];
      if (bias) y += bias[gm];
      if (bnG)  y = y * (bnG[gm] * KBN) + bnB[gm];
      if (doElu) y = elu1(y);
      long idx = (long)bz * sCb + (long)img * M * npi + (long)gm * npi + pos;
      if (res) y += res[idx];
      C[idx] = y;
      if (Cbf) Cbf[idx] = (__bf16)y;
    }
  }
}

// ---------------------------------------------------------------------------
// Block 1: direct conv (cin=1, k=3x2, stride (2,1), pad (0,1), drop last col)
// fused BN + ELU; too small for WMMA (K=6).
// ---------------------------------------------------------------------------
__global__ __launch_bounds__(256) void conv1_kernel(
    const float* __restrict__ x, const float* __restrict__ w,
    const float* __restrict__ bias, const float* __restrict__ g,
    const float* __restrict__ be, float* __restrict__ out,
    __bf16* __restrict__ outbf)
{
  long i = (long)blockIdx.x * 256 + threadIdx.x;
  const long total = 16L * 16 * 128 * WFULL;
  if (i >= total) return;
  int ow = (int)(i % WFULL); long r = i / WFULL;
  int oh = (int)(r % 128); r /= 128;
  int co = (int)(r % 16);  int b = (int)(r / 16);
  const float* xb = x + (long)b * 257 * WFULL;
  float acc = bias[co];
#pragma unroll
  for (int kh = 0; kh < 3; ++kh) {
    int ih = oh * 2 + kh;
#pragma unroll
    for (int kw = 0; kw < 2; ++kw) {
      int iw = ow + kw - 1;
      if (iw >= 0 && iw < WFULL)
        acc += xb[ih * WFULL + iw] * w[(co * 3 + kh) * 2 + kw];
    }
  }
  float y = acc * (g[co] * KBN) + be[co];
  y = elu1(y);
  out[i] = y;
  outbf[i] = (__bf16)y;
}

// ---------------------------------------------------------------------------
// Deformable bilinear gather. Offsets tensor is NCHW [B,2C,H,W]; the torch
// .view semantics mean plane p=b*C+c reads off[p*2HW + (h*W+w)*2 + {0,1}].
// ---------------------------------------------------------------------------
__global__ __launch_bounds__(256) void deform_kernel(
    const float* __restrict__ act, const float* __restrict__ off,
    float* __restrict__ outF, __bf16* __restrict__ outB,
    int BC, int H, int W)
{
  long i = (long)blockIdx.x * 256 + threadIdx.x;
  long HW = (long)H * W;
  long total = (long)BC * HW;
  if (i >= total) return;
  int p = (int)(i / HW); int q = (int)(i - (long)p * HW);
  int h = q / W, w = q - (q / W) * W;
  const float* ob = off + (long)p * 2 * HW + (long)q * 2;
  float ci = fminf(fmaxf(ob[0] + (float)h, 0.f), (float)(H - 1));
  float cj = fminf(fmaxf(ob[1] + (float)w, 0.f), (float)(W - 1));
  float i0 = floorf(ci), i1 = ceilf(ci);
  float j0 = floorf(cj), j1 = ceilf(cj);
  float wi = ci - i0, wj = cj - j0;
  const float* xp = act + (long)p * HW;
  int ii0 = (int)i0, ii1 = (int)i1, jj0 = (int)j0, jj1 = (int)j1;
  float vlt = xp[ii0 * W + jj0];
  float vrt = xp[ii1 * W + jj0];
  float vlb = xp[ii0 * W + jj1];
  float vrb = xp[ii1 * W + jj1];
  float vt = vlt + (vrt - vlt) * wi;
  float vb = vlb + (vrb - vlb) * wi;
  float o  = vt + (vb - vt) * wj;
  outF[i] = o;
  outB[i] = (__bf16)o;
}

__global__ __launch_bounds__(256) void cvt_bf16_kernel(
    const float* __restrict__ s, __bf16* __restrict__ d, long n)
{
  long i = (long)blockIdx.x * 256 + threadIdx.x;
  if (i < n) d[i] = (__bf16)s[i];
}

// convert + zero-pad (pad region must be 0 for the async weight streaming)
__global__ __launch_bounds__(256) void cvt_pad_bf16_kernel(
    const float* __restrict__ s, __bf16* __restrict__ d, long n, long npad)
{
  long i = (long)blockIdx.x * 256 + threadIdx.x;
  if (i < npad) d[i] = (i < n) ? (__bf16)s[i] : (__bf16)0.f;
}

// mean over (H,W) per (b,c): one block per plane
__global__ __launch_bounds__(256) void mean_hw_kernel(
    const float* __restrict__ a, float* __restrict__ out, int HW)
{
  int p = blockIdx.x;
  const float* src = a + (long)p * HW;
  float s = 0.f;
  for (int i = threadIdx.x; i < HW; i += 256) s += src[i];
  __shared__ float red[256];
  red[threadIdx.x] = s; __syncthreads();
  for (int st = 128; st > 0; st >>= 1) {
    if (threadIdx.x < st) red[threadIdx.x] += red[threadIdx.x + st];
    __syncthreads();
  }
  if (threadIdx.x == 0) out[p] = red[0] / (float)HW;
}

// mean over (C,H) per (b,w): one block per column
__global__ __launch_bounds__(256) void mean_ch_kernel(
    const float* __restrict__ a, float* __restrict__ out, int C, int H, int W)
{
  int b = blockIdx.x / W, w = blockIdx.x - (blockIdx.x / W) * W;
  int CH = C * H;
  float s = 0.f;
  for (int i = threadIdx.x; i < CH; i += 256) {
    int c = i / H, h = i - (i / H) * H;
    s += a[((long)(b * C + c) * H + h) * W + w];
  }
  __shared__ float red[256];
  red[threadIdx.x] = s; __syncthreads();
  for (int st = 128; st > 0; st >>= 1) {
    if (threadIdx.x < st) red[threadIdx.x] += red[threadIdx.x + st];
    __syncthreads();
  }
  if (threadIdx.x == 0) out[blockIdx.x] = red[0] / (float)CH;
}

// q[b,o] = BN(W[o,:] . v[b,:] + bias)   (16 x n outputs, n<=256)
__global__ __launch_bounds__(256) void matvec_bn_kernel(
    const float* __restrict__ Wm, const float* __restrict__ bias,
    const float* __restrict__ g, const float* __restrict__ be,
    const float* __restrict__ v, float* __restrict__ out, int n)
{
  int i = blockIdx.x * 256 + threadIdx.x;
  if (i >= 16 * n) return;
  int b = i / n, o = i - (i / n) * n;
  const float* vr = v + (long)b * n;
  const float* wr = Wm + (long)o * n;
  float s = bias[o];
  for (int c = 0; c < n; ++c) s += wr[c] * vr[c];
  out[i] = s * (g[o] * KBN) + be[o];
}

// attn[b,i,j] = softmax_j(q[b,i]*q[b,j]); one block per row, n<=256
__global__ __launch_bounds__(256) void softmax_outer_kernel(
    const float* __restrict__ q, __bf16* __restrict__ attn, int n)
{
  int row = blockIdx.x;           // row = b*n + i
  int b = row / n;
  int j = threadIdx.x;
  float qi = q[row];
  float val = (j < n) ? qi * q[b * n + j] : -3.4e38f;
  __shared__ float red[256];
  red[j] = val; __syncthreads();
  for (int st = 128; st > 0; st >>= 1) {
    if (j < st) red[j] = fmaxf(red[j], red[j + st]);
    __syncthreads();
  }
  float mx = red[0]; __syncthreads();
  float e = (j < n) ? expf(val - mx) : 0.f;
  red[j] = e; __syncthreads();
  for (int st = 128; st > 0; st >>= 1) {
    if (j < st) red[j] += red[j + st];
    __syncthreads();
  }
  float inv = 1.f / red[0];
  if (j < n) attn[(long)row * n + j] = (__bf16)(e * inv);
}

// ---------------------------------------------------------------------------
extern "C" void kernel_launch(void* const* d_in, const int* in_sizes, int n_in,
                              void* d_out, int out_size, void* d_ws, size_t ws_size,
                              hipStream_t stream)
{
  (void)in_sizes; (void)n_in; (void)out_size; (void)ws_size;
  // ----- input map (setup_inputs insertion order) -----
  const float* x = (const float*)d_in[0];
  auto blkBase = [](int li) { return li == 0 ? 1 : 5 * li; };
  auto Wc  = [&](int li) { return (const float*)d_in[blkBase(li) + 0]; };
  auto Bc  = [&](int li) { return (const float*)d_in[blkBase(li) + 1]; };
  auto Gc  = [&](int li) { return (const float*)d_in[blkBase(li) + 2]; };
  auto BEc = [&](int li) { return (const float*)d_in[blkBase(li) + 3]; };
  auto Woc = [&](int li) { return (const float*)d_in[blkBase(li) + 4]; };  // li>=1
  const float* g_w  = (const float*)d_in[25]; const float* g_b  = (const float*)d_in[26];
  const float* g_g  = (const float*)d_in[27]; const float* g_be = (const float*)d_in[28];
  const float* a_w  = (const float*)d_in[29]; const float* a_b  = (const float*)d_in[30];
  const float* a_g  = (const float*)d_in[31]; const float* a_be = (const float*)d_in[32];
  const float* o_w  = (const float*)d_in[33]; const float* o_b  = (const float*)d_in[34];
  const float* o_g  = (const float*)d_in[35]; const float* o_be = (const float*)d_in[36];
  const float* o1_w = (const float*)d_in[37]; const float* o1_b = (const float*)d_in[38];
  const float* o1_g = (const float*)d_in[39]; const float* o1_be= (const float*)d_in[40];

  // ----- workspace carve-out -----
  char* p = (char*)d_ws;
  auto alloc_f = [&](size_t n) { float* r = (float*)p; p += ((n * 4 + 255) & ~(size_t)255); return r; };
  auto alloc_b = [&](size_t n) { __bf16* r = (__bf16*)p; p += ((n * 2 + 255) & ~(size_t)255); return r; };
  const size_t A1 = 8224768, A5 = 7196672, OFFMAX = 16449536;
  float*  actA_f = alloc_f(A1);  __bf16* actA_b = alloc_b(A1);
  float*  actB_f = alloc_f(A1);  __bf16* actB_b = alloc_b(A1);
  float*  off_f  = alloc_f(OFFMAX);
  float*  def_f  = alloc_f(A1);  __bf16* def_b  = alloc_b(A1);
  float*  act5_f = alloc_f(A5);  __bf16* act5_b = alloc_b(A5);
  float*  gsc_f  = alloc_f(A5);
  __bf16* v_b    = alloc_b(A5);
  __bf16* vt_b   = alloc_b(A5);
  __bf16* vc_b   = alloc_b(A5);
  __bf16* attc_b = alloc_b((size_t)16 * 256 * 256);
  __bf16* attt_b = alloc_b((size_t)16 * 251 * 251);
  float*  xc = alloc_f(16 * 256); float* xt = alloc_f(16 * 251);
  float*  qc = alloc_f(16 * 256); float* qt = alloc_f(16 * 251);

  // padded weight buffers (rows rounded to 64-tile for async streaming)
  struct Lyr { int cin, H, Ho, cout; };
  const Lyr ls[4] = {{16,128,63,32},{32,63,31,64},{64,31,15,128},{128,15,7,256}};
  __bf16* wmain_b[4]; __bf16* woff_b[4];
  size_t wmain_n[4], wmain_np[4], woff_n[4], woff_np[4];
  for (int i = 0; i < 4; ++i) {
    int Km = ls[i].cin * 6, Mm = ls[i].cout, Mmp = (Mm + 63) & ~63;
    int Ko = ls[i].cin * 9, Mo = 2 * ls[i].cin, Mop = (Mo + 63) & ~63;
    wmain_n[i] = (size_t)Mm * Km;  wmain_np[i] = (size_t)Mmp * Km + 64;
    woff_n[i]  = (size_t)Mo * Ko;  woff_np[i]  = (size_t)Mop * Ko + 64;
    wmain_b[i] = alloc_b(wmain_np[i]);
    woff_b[i]  = alloc_b(woff_np[i]);
  }
  __bf16* ow_b  = alloc_b(65536);
  __bf16* o1w_b = alloc_b(65536);

  auto cvt = [&](const float* s, __bf16* d, size_t n) {
    cvt_bf16_kernel<<<dim3((unsigned)((n + 255) / 256)), 256, 0, stream>>>(s, d, (long)n);
  };
  auto cvtp = [&](const float* s, __bf16* d, size_t n, size_t np) {
    cvt_pad_bf16_kernel<<<dim3((unsigned)((np + 255) / 256)), 256, 0, stream>>>(
        s, d, (long)n, (long)np);
  };
  for (int i = 0; i < 4; ++i) {
    cvtp(Wc(i + 1),  wmain_b[i], wmain_n[i], wmain_np[i]);
    cvtp(Woc(i + 1), woff_b[i],  woff_n[i],  woff_np[i]);
  }
  cvt(o_w, ow_b, 65536); cvt(o1_w, o1w_b, 65536);

  // ----- block 1 (direct) -----
  conv1_kernel<<<dim3((unsigned)((A1 + 255) / 256)), 256, 0, stream>>>(
      x, Wc(0), Bc(0), Gc(0), BEc(0), actA_f, actA_b);

  // ----- blocks 2..5 (deform -> im2col WMMA GEMM, fused bias/BN/ELU) -----
  float* curF = actA_f; __bf16* curB = actA_b;
  for (int i = 0; i < 4; ++i) {
    const Lyr& l = ls[i];
    const int li = i + 1;
    // offset conv: M=2*cin, K=cin*9, N=16*H*W, 3x3 s1 p1, no epilogue
    {
      int M = 2 * l.cin, K = l.cin * 9, N = 16 * l.H * WFULL;
      dim3 grid((N + 127) / 128, (M + 63) / 64, 1);
      gemm_bf16<1, false><<<grid, 256, 0, stream>>>(
          woff_b[i], curB, off_f, nullptr, M, N, K, K, 0,
          0, 0, 0, l.H * WFULL, nullptr, nullptr, nullptr, 0, nullptr,
          l.cin, l.H, WFULL, l.H, WFULL, 3, 3, 1, 1, 1);
    }
    // bilinear gather
    {
      long total = (long)16 * l.cin * l.H * WFULL;
      deform_kernel<<<dim3((unsigned)((total + 255) / 256)), 256, 0, stream>>>(
          curF, off_f, def_f, def_b, 16 * l.cin, l.H, WFULL);
    }
    // main conv: M=cout, K=cin*6, N=16*Ho*W, 3x2 s(2,1) p(0,1), bias+BN+ELU
    float*  outF = (i == 3) ? act5_f : ((curF == actA_f) ? actB_f : actA_f);
    __bf16* outB = (i == 3) ? act5_b : ((curB == actA_b) ? actB_b : actA_b);
    {
      int M = l.cout, K = l.cin * 6, N = 16 * l.Ho * WFULL;
      dim3 grid((N + 127) / 128, (M + 63) / 64, 1);
      gemm_bf16<1, false><<<grid, 256, 0, stream>>>(
          wmain_b[i], def_b, outF, outB, M, N, K, K, 0,
          0, 0, 0, l.Ho * WFULL, Bc(li), Gc(li), BEc(li), 1, nullptr,
          l.cin, l.H, WFULL, l.Ho, WFULL, 3, 2, 2, 0, 1);
    }
    curF = outF; curB = outB;
  }

  // ----- attention head -----
  const int HW5 = 7 * WFULL;          // 1757
  const long CHW5 = 256L * HW5;       // 449792
  mean_hw_kernel<<<dim3(16 * 256), 256, 0, stream>>>(act5_f, xc, HW5);
  mean_ch_kernel<<<dim3(16 * WFULL), 256, 0, stream>>>(act5_f, xt, 256, 7, WFULL);
  matvec_bn_kernel<<<dim3(16), 256, 0, stream>>>(g_w, g_b, g_g, g_be, xc, qc, 256);
  matvec_bn_kernel<<<dim3(16), 256, 0, stream>>>(a_w, a_b, a_g, a_be, xt, qt, 251);
  softmax_outer_kernel<<<dim3(16 * 256), 256, 0, stream>>>(qc, attc_b, 256);
  softmax_outer_kernel<<<dim3(16 * 251), 256, 0, stream>>>(qt, attt_b, 251);

  // v = BN(conv1x1(act5)) : batched 256x256 x 1757
  {
    dim3 grid((HW5 + 127) / 128, 4, 16);
    gemm_bf16<0, false><<<grid, 256, 0, stream>>>(
        ow_b, act5_b, gsc_f, v_b, 256, HW5, 256, 256, HW5,
        0, CHW5, CHW5, HW5, o_b, o_g, o_be, 0, nullptr,
        0, 0, 0, 0, 0, 1, 1, 1, 0, 0);
  }
  // v_t = v @ atten_t : batched 1792x251 x 251
  {
    dim3 grid((251 + 127) / 128, (1792 + 63) / 64, 16);
    gemm_bf16<0, false><<<grid, 256, 0, stream>>>(
        v_b, attt_b, gsc_f, vt_b, 1792, 251, 251, 251, 251,
        CHW5, 251L * 251, CHW5, 251, nullptr, nullptr, nullptr, 0, nullptr,
        0, 0, 0, 0, 0, 1, 1, 1, 0, 0);
  }
  // v_c[c2,hw] = sum_c atten_c[c,c2] * v_t[c,hw] : A transposed, batched
  {
    dim3 grid((HW5 + 127) / 128, 4, 16);
    gemm_bf16<0, true><<<grid, 256, 0, stream>>>(
        attc_b, vt_b, gsc_f, vc_b, 256, HW5, 256, 256, HW5,
        256L * 256, CHW5, CHW5, HW5, nullptr, nullptr, nullptr, 0, nullptr,
        0, 0, 0, 0, 0, 1, 1, 1, 0, 0);
  }
  // out = BN(conv1x1(v_c)) + act5  (written straight to d_out)
  {
    dim3 grid((HW5 + 127) / 128, 4, 16);
    gemm_bf16<0, false><<<grid, 256, 0, stream>>>(
        o1w_b, vc_b, (float*)d_out, nullptr, 256, HW5, 256, 256, HW5,
        0, CHW5, CHW5, HW5, o1_b, o1_g, o1_be, 0, act5_f,
        0, 0, 0, 0, 0, 1, 1, 1, 0, 0);
  }
}
